// SelfAttention_20529943675099
// MI455X (gfx1250) — compile-verified, matrix operand order checked by reading, not run on hardware
//
#include <hip/hip_runtime.h>
#include <hip/hip_bf16.h>
#include <stdint.h>

// ---------------------------------------------------------------------------
// SelfAttention block (GroupNorm -> QKV 1x1conv -> MHA -> out proj -> +x)
// B=16, C=512, H=W=32 (HW=1024), HEADS=8, HEAD_DIM=64, GROUPS=32
// GEMMs on v_wmma_f32_16x16x32_bf16; tiles staged with gfx1250 async
// global->LDS DMA (ASYNCcnt) and double buffering.
// ---------------------------------------------------------------------------

#define B_   16
#define C_   512
#define HW_  1024
#define HEADS_ 8
#define D_   64
#define GROUPS_ 32
#define CPG_ 16
#define EPS_ 1e-5f
#define SCALE_ 0.125f   // 64^-0.5

typedef __attribute__((ext_vector_type(16))) __bf16 v16bf;
typedef __attribute__((ext_vector_type(8)))  float  v8f;

// ---- async global->LDS copy (16B per lane), tracked by ASYNCcnt -----------
__device__ __forceinline__ void async_b128(void* lds, const void* gaddr) {
    unsigned lo = (unsigned)(uintptr_t)lds;           // LDS aperture: addr[31:0]
    unsigned long long ga = (unsigned long long)(uintptr_t)gaddr;
    asm volatile("global_load_async_to_lds_b128 %0, %1, off"
                 :: "v"(lo), "v"(ga) : "memory");
}
__device__ __forceinline__ void wait_async0() {
    asm volatile("s_wait_asynccnt 0" ::: "memory");
}

// ---- WMMA fragment helper (ISA 7.12.2 16-bit layout, wave32) --------------
// A lane's 16 bf16 fragment elements are two contiguous 8-element runs of its
// row:  [row][h*8 .. h*8+7] and [row][16+h*8 .. 16+h*8+7]   (h = lane>>4).
// Works for A tiles stored [M][K] and B tiles stored [N][K] (N-major).
__device__ __forceinline__ v16bf frag_row(const __bf16* t, int ld, int lane) {
    const int r = lane & 15, h = lane >> 4;
    const __bf16* p = t + r * ld + h * 8;
    union { uint4 u[2]; v16bf v; } u;
    u.u[0] = *(const uint4*)p;
    u.u[1] = *(const uint4*)(p + 16);
    return u.v;
}

__device__ __forceinline__ v8f wmma_bf16(v16bf a, v16bf b, v8f c) {
    return __builtin_amdgcn_wmma_f32_16x16x32_bf16(
        /*neg_a=*/false, a, /*neg_b=*/false, b,
        /*c_mod=*/(short)0, c, /*reuse_a=*/false, /*reuse_b=*/false);
}

// ---------------------------------------------------------------------------
// Kernel 1: GroupNorm -> bf16 xn  (one block per (b, group); data contiguous)
// ---------------------------------------------------------------------------
__global__ __launch_bounds__(256) void gn_kernel(const float* __restrict__ x,
                                                 const float* __restrict__ gamma,
                                                 const float* __restrict__ beta,
                                                 __bf16* __restrict__ xn) {
    const int b = blockIdx.x >> 5, g = blockIdx.x & 31;
    const int N = CPG_ * HW_;  // 16384 contiguous floats
    const float* xp = x + ((size_t)b * C_ + g * CPG_) * HW_;
    float s = 0.f, s2 = 0.f;
    for (int i = threadIdx.x; i < N; i += 256) {
        float v = xp[i];
        s += v; s2 += v * v;
    }
#pragma unroll
    for (int m = 16; m; m >>= 1) {
        s  += __shfl_xor(s,  m, 32);
        s2 += __shfl_xor(s2, m, 32);
    }
    __shared__ float red[16];
    const int lane = threadIdx.x & 31, wave = threadIdx.x >> 5;
    if (lane == 0) { red[wave] = s; red[8 + wave] = s2; }
    __syncthreads();
    if (wave == 0) {
        float a = (lane < 8) ? red[lane] : 0.f;
        float c = (lane < 8) ? red[8 + lane] : 0.f;
#pragma unroll
        for (int m = 4; m; m >>= 1) {
            a += __shfl_xor(a, m, 32);
            c += __shfl_xor(c, m, 32);
        }
        if (lane == 0) { red[0] = a; red[1] = c; }
    }
    __syncthreads();
    const float mu   = red[0] / (float)N;
    const float var  = red[1] / (float)N - mu * mu;
    const float rinv = rsqrtf(var + EPS_);
    __bf16* op = xn + ((size_t)b * C_ + g * CPG_) * HW_;
    for (int i = threadIdx.x; i < N; i += 256) {
        int c = g * CPG_ + (i >> 10);
        op[i] = (__bf16)(((xp[i] - mu) * rinv) * gamma[c] + beta[c]);
    }
}

// ---------------------------------------------------------------------------
// Kernel 2: f32 -> bf16 weight conversion
// ---------------------------------------------------------------------------
__global__ __launch_bounds__(256) void cvt_kernel(const float* __restrict__ w,
                                                  __bf16* __restrict__ o, int n) {
    int i = blockIdx.x * 256 + threadIdx.x;
    if (i < n) o[i] = (__bf16)w[i];
}

// ---------------------------------------------------------------------------
// Kernel 3: QKV GEMM  (1536 x 512) x (512 x 1024) per batch, WMMA bf16
// 128x128 block tile, double-buffered; A via async DMA, B transposed via VGPR.
// Scatters to Q [b][h][p][d], K [b][h][p][d], V [b][h][d][p].
// ---------------------------------------------------------------------------
__global__ __launch_bounds__(256) void qkv_gemm(const __bf16* __restrict__ xn,
                                                const __bf16* __restrict__ wq,
                                                const float* __restrict__ bqkv,
                                                __bf16* __restrict__ qw,
                                                __bf16* __restrict__ kw,
                                                __bf16* __restrict__ vw) {
    const int b  = blockIdx.z;
    const int m0 = blockIdx.y * 128;   // over 1536 outputs
    const int n0 = blockIdx.x * 128;   // over 1024 positions
    __shared__ __align__(16) __bf16 As[2][128 * 32];   // [m][k]
    __shared__ __align__(16) __bf16 Bs[2][128 * 32];   // [n][k]  (N-major)
    const int tid = threadIdx.x, lane = tid & 31, wave = tid >> 5;
    const int wm = wave & 3, wn = wave >> 2;   // 4x2 wave grid -> 32x64 per wave
    v8f acc[2][4] = {};
    const __bf16* xb = xn + (size_t)b * C_ * HW_;

    auto stage = [&](int k0, int bufi) {
        // A tile: rows contiguous in k -> async DMA straight into LDS
        for (int off = tid * 8; off < 128 * 32; off += 256 * 8) {
            int r = off >> 5, c = off & 31;
            async_b128(&As[bufi][off], &wq[(size_t)(m0 + r) * C_ + k0 + c]);
        }
        // B tile: xn is [k][p]; transpose into Bs[n][k] (coalesced global reads)
        for (int off = tid * 8; off < 128 * 32; off += 256 * 8) {
            int kk = off >> 7, nn = off & 127;
            __bf16 tmp[8];
            *(uint4*)tmp = *(const uint4*)&xb[(size_t)(k0 + kk) * HW_ + n0 + nn];
#pragma unroll
            for (int j = 0; j < 8; ++j) Bs[bufi][(nn + j) * 32 + kk] = tmp[j];
        }
    };

    stage(0, 0);
    wait_async0();
    __syncthreads();
    int buf = 0;
    for (int k0 = 0; k0 < C_; k0 += 32) {
        if (k0 + 32 < C_) stage(k0 + 32, buf ^ 1);
#pragma unroll
        for (int i = 0; i < 2; ++i) {
            v16bf a = frag_row(&As[buf][(wm * 32 + i * 16) * 32], 32, lane);
#pragma unroll
            for (int j = 0; j < 4; ++j) {
                v16bf bb = frag_row(&Bs[buf][(wn * 64 + j * 16) * 32], 32, lane);
                acc[i][j] = wmma_bf16(a, bb, acc[i][j]);
            }
        }
        wait_async0();
        __syncthreads();
        buf ^= 1;
    }
    // epilogue: bias + scatter into attention layouts
#pragma unroll
    for (int i = 0; i < 2; ++i)
#pragma unroll
        for (int j = 0; j < 4; ++j)
#pragma unroll
            for (int r = 0; r < 8; ++r) {
                int o = m0 + wm * 32 + i * 16 + r + (lane >> 4) * 8;
                int p = n0 + wn * 64 + j * 16 + (lane & 15);
                float val = acc[i][j][r] + bqkv[o];
                int t = o >> 9, cc = o & 511, hh = cc >> 6, dd = cc & 63;
                size_t bh = (size_t)b * HEADS_ + hh;
                if (t == 0)      qw[(bh * HW_ + p) * D_ + dd] = (__bf16)val;
                else if (t == 1) kw[(bh * HW_ + p) * D_ + dd] = (__bf16)val;
                else             vw[(bh * D_ + dd) * HW_ + p] = (__bf16)val;
            }
}

// ---------------------------------------------------------------------------
// Kernel 4: flash attention per (b, head, 64-query tile); 4 waves x 16 queries
// S = Q K^T (scaled), online softmax, O += P V.  Double-buffered async K/V.
// Q,K global layout [p][d]; V global layout [d][p].
// ---------------------------------------------------------------------------
__global__ __launch_bounds__(128) void attn_kernel(const __bf16* __restrict__ qw,
                                                   const __bf16* __restrict__ kw,
                                                   const __bf16* __restrict__ vw,
                                                   __bf16* __restrict__ ob) {
    const int b = blockIdx.z, h = blockIdx.y, q0 = blockIdx.x * 64;
    const int tid = threadIdx.x, lane = tid & 31, wave = tid >> 5;
    __shared__ __align__(16) __bf16 Kt[2][32 * 64];   // [key][d]  (N-major, QK)
    __shared__ __align__(16) __bf16 Vt[2][64 * 32];   // [d][key]  (N-major, PV)
    __shared__ __align__(16) __bf16 Pt[4][16 * 32];   // per-wave P staging

    const size_t bh = (size_t)b * HEADS_ + h;
    const __bf16* Q  = qw + (bh * HW_ + q0 + wave * 16) * D_;  // wave's [16][64]
    const __bf16* Kb = kw + bh * HW_ * D_;                     // [p][d]
    const __bf16* Vb = vw + bh * D_ * HW_;                     // [d][p]

    // Q A-fragments straight from global (b128 loads), two K=32 halves of d=64
    v16bf aq[2];
#pragma unroll
    for (int kt = 0; kt < 2; ++kt) aq[kt] = frag_row(Q + kt * 32, D_, lane);

    auto stage_kv = [&](int mt, int bufi) {
        for (int off = tid * 8; off < 32 * 64; off += 128 * 8) {
            int p = off >> 6, c = off & 63;
            async_b128(&Kt[bufi][off], &Kb[(size_t)(mt + p) * D_ + c]);
        }
        for (int off = tid * 8; off < 64 * 32; off += 128 * 8) {
            int d = off >> 5, c = off & 31;
            async_b128(&Vt[bufi][off], &Vb[(size_t)d * HW_ + mt + c]);
        }
    };

    v8f o_acc[4] = {};
    float run_m[8], run_l[8];
#pragma unroll
    for (int r = 0; r < 8; ++r) { run_m[r] = -1e30f; run_l[r] = 0.f; }

    stage_kv(0, 0);
    wait_async0();
    __syncthreads();
    int buf = 0;
    for (int mt = 0; mt < HW_; mt += 32) {
        if (mt + 32 < HW_) stage_kv(mt + 32, buf ^ 1);  // overlap DMA w/ compute

        // S tile: 16 queries x 32 keys (two 16-key N-blocks, K=64 over d)
        v8f s0 = {}, s1 = {};
#pragma unroll
        for (int kt = 0; kt < 2; ++kt) {
            s0 = wmma_bf16(aq[kt], frag_row(&Kt[buf][0 * 64]  + kt * 32, 64, lane), s0);
            s1 = wmma_bf16(aq[kt], frag_row(&Kt[buf][16 * 64] + kt * 32, 64, lane), s1);
        }

        // online softmax (row reductions stay inside the 16-lane half)
#pragma unroll
        for (int r = 0; r < 8; ++r) {
            float v0 = s0[r] * SCALE_, v1 = s1[r] * SCALE_;
            float mx = fmaxf(v0, v1);
#pragma unroll
            for (int m = 8; m; m >>= 1) mx = fmaxf(mx, __shfl_xor(mx, m, 32));
            float nm = fmaxf(run_m[r], mx);
            float alpha = __expf(run_m[r] - nm);
            float p0 = __expf(v0 - nm), p1 = __expf(v1 - nm);
            float rs = p0 + p1;
#pragma unroll
            for (int m = 8; m; m >>= 1) rs += __shfl_xor(rs, m, 32);
            run_l[r] = run_l[r] * alpha + rs;
            run_m[r] = nm;
#pragma unroll
            for (int dn = 0; dn < 4; ++dn) o_acc[dn][r] *= alpha;
            int row = r + (lane >> 4) * 8, col = lane & 15;
            Pt[wave][row * 32 + col]      = (__bf16)p0;
            Pt[wave][row * 32 + 16 + col] = (__bf16)p1;
        }
        // Pt is wave-private and LDS is in-order per wave: no block barrier.

        // O += P(16x32) * V(32x64)
        v16bf ap = frag_row(&Pt[wave][0], 32, lane);
#pragma unroll
        for (int dn = 0; dn < 4; ++dn)
            o_acc[dn] = wmma_bf16(ap, frag_row(&Vt[buf][(dn * 16) * 32], 32, lane), o_acc[dn]);

        wait_async0();     // next tile's DMA has landed
        __syncthreads();   // everyone done reading buf before it is re-staged
        buf ^= 1;
    }

    // normalize + store attention output as [b][p][c] bf16
#pragma unroll
    for (int dn = 0; dn < 4; ++dn)
#pragma unroll
        for (int r = 0; r < 8; ++r) {
            int row = r + (lane >> 4) * 8;
            int p = q0 + wave * 16 + row;
            int cchan = h * D_ + dn * 16 + (lane & 15);
            float val = o_acc[dn][r] / run_l[r];
            ob[((size_t)b * HW_ + p) * C_ + cchan] = (__bf16)val;
        }
}

// ---------------------------------------------------------------------------
// Kernel 5: out projection (512x512)x(512x1024) + bias + residual, WMMA bf16
// Both operands N/M-major in memory -> both tiles via async DMA, double-buffered.
// ---------------------------------------------------------------------------
__global__ __launch_bounds__(256) void out_gemm(const __bf16* __restrict__ ob,
                                                const __bf16* __restrict__ wo,
                                                const float* __restrict__ bout,
                                                const float* __restrict__ x,
                                                float* __restrict__ out) {
    const int b  = blockIdx.z;
    const int m0 = blockIdx.y * 128;   // over 512 outputs
    const int n0 = blockIdx.x * 128;   // over 1024 positions
    __shared__ __align__(16) __bf16 As[2][128 * 32];   // [m][k]
    __shared__ __align__(16) __bf16 Bs[2][128 * 32];   // [n][k]
    const int tid = threadIdx.x, lane = tid & 31, wave = tid >> 5;
    const int wm = wave & 3, wn = wave >> 2;
    v8f acc[2][4] = {};

    auto stage = [&](int k0, int bufi) {
        for (int off = tid * 8; off < 128 * 32; off += 256 * 8) {
            int r = off >> 5, c = off & 31;
            async_b128(&As[bufi][off], &wo[(size_t)(m0 + r) * C_ + k0 + c]);
        }
        for (int off = tid * 8; off < 128 * 32; off += 256 * 8) {
            int nn = off >> 5, kk = off & 31;
            async_b128(&Bs[bufi][off], &ob[((size_t)b * HW_ + n0 + nn) * C_ + k0 + kk]);
        }
    };

    stage(0, 0);
    wait_async0();
    __syncthreads();
    int buf = 0;
    for (int k0 = 0; k0 < C_; k0 += 32) {
        if (k0 + 32 < C_) stage(k0 + 32, buf ^ 1);
#pragma unroll
        for (int i = 0; i < 2; ++i) {
            v16bf a = frag_row(&As[buf][(wm * 32 + i * 16) * 32], 32, lane);
#pragma unroll
            for (int j = 0; j < 4; ++j) {
                v16bf bb = frag_row(&Bs[buf][(wn * 64 + j * 16) * 32], 32, lane);
                acc[i][j] = wmma_bf16(a, bb, acc[i][j]);
            }
        }
        wait_async0();
        __syncthreads();
        buf ^= 1;
    }
#pragma unroll
    for (int i = 0; i < 2; ++i)
#pragma unroll
        for (int j = 0; j < 4; ++j)
#pragma unroll
            for (int r = 0; r < 8; ++r) {
                int o = m0 + wm * 32 + i * 16 + r + (lane >> 4) * 8;
                int p = n0 + wn * 64 + j * 16 + (lane & 15);
                size_t idx = ((size_t)b * C_ + o) * HW_ + p;
                out[idx] = acc[i][j][r] + bout[o] + x[idx];
            }
}

// ---------------------------------------------------------------------------
extern "C" void kernel_launch(void* const* d_in, const int* in_sizes, int n_in,
                              void* d_out, int out_size, void* d_ws, size_t ws_size,
                              hipStream_t stream) {
    const float* x     = (const float*)d_in[0];
    const float* gamma = (const float*)d_in[1];
    const float* beta  = (const float*)d_in[2];
    const float* wqkv  = (const float*)d_in[3];
    const float* bqkv  = (const float*)d_in[4];
    const float* wout  = (const float*)d_in[5];
    const float* bout  = (const float*)d_in[6];
    float* out = (float*)d_out;

    char* ws = (char*)d_ws;
    // workspace layout (bytes)
    __bf16* xn = (__bf16*)(ws);                      // 16 MB (reused as attn out)
    __bf16* wq = (__bf16*)(ws + 16777216);           // 1.5 MB
    __bf16* wo = (__bf16*)(ws + 18350080);           // 0.5 MB
    __bf16* qb = (__bf16*)(ws + 18874368);           // 16 MB  Q [b][h][p][d]
    __bf16* kb = (__bf16*)(ws + 35651584);           // 16 MB  K [b][h][p][d]
    __bf16* vb = (__bf16*)(ws + 52428800);           // 16 MB  V [b][h][d][p]
    __bf16* ob = xn;                                 // reuse after QKV GEMM

    gn_kernel<<<B_ * GROUPS_, 256, 0, stream>>>(x, gamma, beta, xn);
    cvt_kernel<<<(3 * C_ * C_ + 255) / 256, 256, 0, stream>>>(wqkv, wq, 3 * C_ * C_);
    cvt_kernel<<<(C_ * C_ + 255) / 256, 256, 0, stream>>>(wout, wo, C_ * C_);
    qkv_gemm<<<dim3(HW_ / 128, 3 * C_ / 128, B_), 256, 0, stream>>>(xn, wq, bqkv, qb, kb, vb);
    attn_kernel<<<dim3(HW_ / 64, HEADS_, B_), 128, 0, stream>>>(qb, kb, vb, ob);
    out_gemm<<<dim3(HW_ / 128, C_ / 128, B_), 256, 0, stream>>>(ob, wo, bout, x, out);
}
